// SpatialAttention_40037685133393
// MI455X (gfx1250) — matrix operand order repaired and checked by reading, not compiled, not run on hardware
//
#include <hip/hip_runtime.h>
#include <hip/hip_bf16.h>

typedef __attribute__((ext_vector_type(16))) _Float16 v16h;
typedef __attribute__((ext_vector_type(8)))  _Float16 v8h;
typedef __attribute__((ext_vector_type(4)))  _Float16 v4h;
typedef __attribute__((ext_vector_type(8)))  float    v8f;

#define NNODES 4096
#define HID    512
#define NHEADS 8
#define HD     64
#define MASKW  128   // 4096 bits / 32

static __device__ inline v8h ldv8(const _Float16* p) { return *(const v8h*)p; }

static __device__ inline v16h mkfrag(v8h lo, v8h hi) {
    union { v16h v; v8h h[2]; } u;
    u.h[0] = lo; u.h[1] = hi;
    return u.v;
}

static __device__ inline v8f wmma16(v16h a, v16h b, v8f c) {
    return __builtin_amdgcn_wmma_f32_16x16x32_f16(false, a, false, b, (short)0, c,
                                                  false, false);
}

// ---------------------------------------------------------------- convert f32 -> f16
__global__ void cvt_f16_kernel(const float* __restrict__ s, _Float16* __restrict__ d, int n4) {
    int i = blockIdx.x * blockDim.x + threadIdx.x;
    if (i < n4) {
        float4 v = ((const float4*)s)[i];
        v4h o = { (_Float16)v.x, (_Float16)v.y, (_Float16)v.z, (_Float16)v.w };
        ((v4h*)d)[i] = o;
    }
}

// ---------------------------------------------------------------- adjacency bitmask
__global__ void mask_zero_kernel(unsigned* __restrict__ m, int nwords) {
    int i = blockIdx.x * blockDim.x + threadIdx.x;
    if (i < nwords) m[i] = 0u;
}

__global__ void mask_build_kernel(const int* __restrict__ ei, int E, unsigned* __restrict__ m) {
    int e = blockIdx.x * blockDim.x + threadIdx.x;
    if (e < E) {
        int r = ei[e];         // query row  (edge_index[0])
        int c = ei[E + e];     // key column (edge_index[1])
        atomicOr(&m[r * MASKW + (c >> 5)], 1u << (c & 31));
    }
}

// ---------------------------------------------------------------- QKV projection (WMMA)
// grid (256, 2, 3); block 128 (4 waves). Wave: 16 rows x 64 cols, K-loop 512/32.
__global__ __launch_bounds__(128)
void qkv_kernel(const _Float16* __restrict__ xh,      // [4096][512]
                const _Float16* __restrict__ Whall,   // [3][512][512] q,k,v
                const float* __restrict__ bq, const float* __restrict__ bk,
                const float* __restrict__ bv,
                _Float16* __restrict__ Qh,            // [8][4096][64]  (pre-scaled 1/8)
                _Float16* __restrict__ Kh,            // [8][4096][64]
                _Float16* __restrict__ Vt)            // [8][64][4096]
{
    int lane = threadIdx.x & 31;
    int wv   = threadIdx.x >> 5;
    int g    = lane >> 4;
    int ln   = lane & 15;
    int sel  = blockIdx.z;
    int n0   = blockIdx.x * 16;
    int ob   = blockIdx.y * 256 + wv * 64;

    const _Float16* W = Whall + (size_t)sel * HID * HID;

    v8f c0 = {}, c1 = {}, c2 = {}, c3 = {};
    const _Float16* arow = xh + (size_t)(n0 + ln) * HID;

    for (int kb = 0; kb < HID; kb += 32) {
        v16h a = mkfrag(ldv8(arow + kb + 16 * g), ldv8(arow + kb + 16 * g + 8));
        const _Float16* w0 = W + (size_t)(ob + 0 * 16 + ln) * HID + kb + 16 * g;
        const _Float16* w1 = W + (size_t)(ob + 1 * 16 + ln) * HID + kb + 16 * g;
        const _Float16* w2 = W + (size_t)(ob + 2 * 16 + ln) * HID + kb + 16 * g;
        const _Float16* w3 = W + (size_t)(ob + 3 * 16 + ln) * HID + kb + 16 * g;
        c0 = wmma16(a, mkfrag(ldv8(w0), ldv8(w0 + 8)), c0);
        c1 = wmma16(a, mkfrag(ldv8(w1), ldv8(w1 + 8)), c1);
        c2 = wmma16(a, mkfrag(ldv8(w2), ldv8(w2 + 8)), c2);
        c3 = wmma16(a, mkfrag(ldv8(w3), ldv8(w3 + 8)), c3);
    }

    v8f acc[4] = { c0, c1, c2, c3 };
    #pragma unroll
    for (int t = 0; t < 4; ++t) {
        int o = ob + t * 16 + ln;                 // output column
        float bias = (sel == 0) ? bq[o] : (sel == 1) ? bk[o] : bv[o];
        if (sel == 2) {
            // Vt[o][n] : 8 consecutive n values -> one 16B store
            v8h p;
            #pragma unroll
            for (int v = 0; v < 8; ++v) p[v] = (_Float16)(acc[t][v] + bias);
            *(v8h*)(Vt + (size_t)o * NNODES + n0 + 8 * g) = p;
        } else {
            int h = o >> 6, d = o & 63;
            #pragma unroll
            for (int v = 0; v < 8; ++v) {
                int n = n0 + v + 8 * g;
                float val = acc[t][v] + bias;
                if (sel == 0) val *= 0.125f;      // fold 1/sqrt(64) into Q
                _Float16* dst = (sel == 0 ? Qh : Kh);
                dst[((size_t)h * NNODES + n) * HD + d] = (_Float16)val;
            }
        }
    }
}

// ---------------------------------------------------------------- flash attention (WMMA)
// grid (64, 8); block 128 (4 waves). Wave: one 16-query tile of one head.
__global__ __launch_bounds__(128)
void attn_kernel(const _Float16* __restrict__ Qh,    // [8][4096][64] pre-scaled
                 const _Float16* __restrict__ Kh,    // [8][4096][64]
                 const _Float16* __restrict__ Vt,    // [8][64][4096]
                 const unsigned* __restrict__ mask,  // [4096][128]
                 _Float16* __restrict__ attO)        // [4096][512]
{
    __shared__ __align__(32) _Float16 lds[4][16 * 32];

    int lane = threadIdx.x & 31;
    int wv   = threadIdx.x >> 5;
    int g    = lane >> 4;
    int ln   = lane & 15;
    int qt   = blockIdx.x * 4 + wv;       // 0..255
    int h    = blockIdx.y;
    int n0   = qt * 16;

    const _Float16* Kbase = Kh + (size_t)h * NNODES * HD;
    const _Float16* Vbase = Vt + (size_t)h * HD * NNODES;

    // Q fragments (d = 0..31 and 32..63), row = n0 + ln
    const _Float16* qr = Qh + ((size_t)h * NNODES + n0 + ln) * HD;
    v16h qa0 = mkfrag(ldv8(qr +  0 + 16 * g), ldv8(qr + 16 * g + 8));
    v16h qa1 = mkfrag(ldv8(qr + 32 + 16 * g), ldv8(qr + 32 + 16 * g + 8));

    v8f oacc[4] = {};
    float mrow[8], lsum[8];
    #pragma unroll
    for (int v = 0; v < 8; ++v) { mrow[v] = -1e30f; lsum[v] = 0.f; }

    for (int j = 0; j < NNODES / 32; ++j) {
        int mb = j * 32;

        unsigned w[8];
        #pragma unroll
        for (int v = 0; v < 8; ++v)
            w[v] = mask[(size_t)(n0 + v + 8 * g) * MASKW + j];

        // score tiles: S0 (keys mb..mb+15), S1 (mb+16..mb+31); contraction d=64
        v8f s0 = {}, s1 = {};
        {
            const _Float16* kr = Kbase + (size_t)(mb + ln) * HD;
            s0 = wmma16(qa0, mkfrag(ldv8(kr + 16 * g), ldv8(kr + 16 * g + 8)), s0);
            s0 = wmma16(qa1, mkfrag(ldv8(kr + 32 + 16 * g), ldv8(kr + 32 + 16 * g + 8)), s0);
        }
        {
            const _Float16* kr = Kbase + (size_t)(mb + 16 + ln) * HD;
            s1 = wmma16(qa0, mkfrag(ldv8(kr + 16 * g), ldv8(kr + 16 * g + 8)), s1);
            s1 = wmma16(qa1, mkfrag(ldv8(kr + 32 + 16 * g), ldv8(kr + 32 + 16 * g + 8)), s1);
        }

        float p0[8], p1[8], alpha[8];
        #pragma unroll
        for (int v = 0; v < 8; ++v) {
            float a0 = ((w[v] >> ln) & 1u)        ? s0[v] : -1e30f;
            float a1 = ((w[v] >> (ln + 16)) & 1u) ? s1[v] : -1e30f;
            // row max over the 32 key columns (2 per lane, 16 lanes of a 16-group)
            float t = fmaxf(a0, a1);
            t = fmaxf(t, __shfl_xor(t, 1, 32));
            t = fmaxf(t, __shfl_xor(t, 2, 32));
            t = fmaxf(t, __shfl_xor(t, 4, 32));
            t = fmaxf(t, __shfl_xor(t, 8, 32));
            float mn = fmaxf(mrow[v], t);
            float al = __expf(mrow[v] - mn);
            mrow[v]  = mn;
            alpha[v] = al;
            float e0 = (a0 > -1e29f) ? __expf(a0 - mn) : 0.f;
            float e1 = (a1 > -1e29f) ? __expf(a1 - mn) : 0.f;
            float rs = e0 + e1;
            rs += __shfl_xor(rs, 1, 32);
            rs += __shfl_xor(rs, 2, 32);
            rs += __shfl_xor(rs, 4, 32);
            rs += __shfl_xor(rs, 8, 32);
            lsum[v] = lsum[v] * al + rs;
            p0[v] = e0; p1[v] = e1;
        }

        #pragma unroll
        for (int t2 = 0; t2 < 4; ++t2)
            #pragma unroll
            for (int v = 0; v < 8; ++v) oacc[t2][v] *= alpha[v];

        // C-layout -> A-layout via per-wave LDS tile [16 q][32 m]
        _Float16* lp = lds[wv];
        #pragma unroll
        for (int v = 0; v < 8; ++v) {
            int r = v + 8 * g;
            lp[r * 32 + ln]      = (_Float16)p0[v];
            lp[r * 32 + 16 + ln] = (_Float16)p1[v];
        }
        asm volatile("" ::: "memory");   // LDS is in-order per wave; just pin ordering
        v16h pa = mkfrag(ldv8(lp + ln * 32 + 16 * g), ldv8(lp + ln * 32 + 16 * g + 8));

        // O += P (16x32) * V (32x16 per d-tile); B from Vt, contiguous in key dim
        #pragma unroll
        for (int t2 = 0; t2 < 4; ++t2) {
            const _Float16* vr = Vbase + (size_t)(t2 * 16 + ln) * NNODES + mb + 16 * g;
            oacc[t2] = wmma16(pa, mkfrag(ldv8(vr), ldv8(vr + 8)), oacc[t2]);
        }
    }

    float inv[8];
    #pragma unroll
    for (int v = 0; v < 8; ++v) inv[v] = 1.f / lsum[v];
    #pragma unroll
    for (int t2 = 0; t2 < 4; ++t2)
        #pragma unroll
        for (int v = 0; v < 8; ++v) {
            int n = n0 + v + 8 * g;
            attO[(size_t)n * HID + h * HD + t2 * 16 + ln] = (_Float16)(oacc[t2][v] * inv[v]);
        }
}

// ---------------------------------------------------------------- output proj + residual
// grid (256, 2); block 128. y = x + attO @ Wo^T + bo   (fp32)
__global__ __launch_bounds__(128)
void oproj_kernel(const _Float16* __restrict__ aO, const _Float16* __restrict__ W,
                  const float* __restrict__ bo, const float* __restrict__ x,
                  float* __restrict__ y)
{
    int lane = threadIdx.x & 31;
    int wv   = threadIdx.x >> 5;
    int g    = lane >> 4;
    int ln   = lane & 15;
    int n0   = blockIdx.x * 16;
    int ob   = blockIdx.y * 256 + wv * 64;

    v8f c[4] = {};
    const _Float16* arow = aO + (size_t)(n0 + ln) * HID;
    for (int kb = 0; kb < HID; kb += 32) {
        v16h a = mkfrag(ldv8(arow + kb + 16 * g), ldv8(arow + kb + 16 * g + 8));
        #pragma unroll
        for (int t = 0; t < 4; ++t) {
            const _Float16* wr = W + (size_t)(ob + t * 16 + ln) * HID + kb + 16 * g;
            c[t] = wmma16(a, mkfrag(ldv8(wr), ldv8(wr + 8)), c[t]);
        }
    }
    #pragma unroll
    for (int t = 0; t < 4; ++t) {
        int o = ob + t * 16 + ln;
        float bias = bo[o];
        #pragma unroll
        for (int v = 0; v < 8; ++v) {
            int n = n0 + v + 8 * g;
            y[(size_t)n * HID + o] = x[(size_t)n * HID + o] + c[t][v] + bias;
        }
    }
}

// ---------------------------------------------------------------- LayerNorm
__global__ __launch_bounds__(128)
void ln_kernel(const float* __restrict__ y, const float* __restrict__ gg,
               const float* __restrict__ bb, float* __restrict__ out)
{
    __shared__ float red[128];
    int row = blockIdx.x, t = threadIdx.x;
    float4 v = ((const float4*)(y + (size_t)row * HID))[t];
    red[t] = v.x + v.y + v.z + v.w;
    __syncthreads();
    for (int o = 64; o > 0; o >>= 1) { if (t < o) red[t] += red[t + o]; __syncthreads(); }
    float mu = red[0] * (1.f / HID);
    __syncthreads();
    float dx = v.x - mu, dy = v.y - mu, dz = v.z - mu, dw = v.w - mu;
    red[t] = dx * dx + dy * dy + dz * dz + dw * dw;
    __syncthreads();
    for (int o = 64; o > 0; o >>= 1) { if (t < o) red[t] += red[t + o]; __syncthreads(); }
    float r = rsqrtf(red[0] * (1.f / HID) + 1e-5f);
    float4 g4 = ((const float4*)gg)[t];
    float4 b4 = ((const float4*)bb)[t];
    float4 o4 = { g4.x * dx * r + b4.x, g4.y * dy * r + b4.y,
                  g4.z * dz * r + b4.z, g4.w * dw * r + b4.w };
    ((float4*)(out + (size_t)row * HID))[t] = o4;
}

// ---------------------------------------------------------------- launch
extern "C" void kernel_launch(void* const* d_in, const int* in_sizes, int n_in,
                              void* d_out, int out_size, void* d_ws, size_t ws_size,
                              hipStream_t stream) {
    const float* x   = (const float*)d_in[0];
    const int*   ei  = (const int*)d_in[1];
    const float* Wq  = (const float*)d_in[2];
    const float* bq  = (const float*)d_in[3];
    const float* Wk  = (const float*)d_in[4];
    const float* bk  = (const float*)d_in[5];
    const float* Wv  = (const float*)d_in[6];
    const float* bv  = (const float*)d_in[7];
    const float* Wo  = (const float*)d_in[8];
    const float* bo  = (const float*)d_in[9];
    const float* lng = (const float*)d_in[10];
    const float* lnb = (const float*)d_in[11];
    int E = in_sizes[1] / 2;

    char* ws = (char*)d_ws;
    const size_t MB = 1u << 20;
    _Float16* xh   = (_Float16*)(ws + 0 * MB);   // 4 MB
    _Float16* Whll = (_Float16*)(ws + 4 * MB);   // 2 MB: Wq,Wk,Wv,Wo (f16)
    _Float16* Qh   = (_Float16*)(ws + 6 * MB);   // 4 MB
    _Float16* Kh   = (_Float16*)(ws + 10 * MB);  // 4 MB
    _Float16* Vt   = (_Float16*)(ws + 14 * MB);  // 4 MB
    _Float16* aO   = (_Float16*)(ws + 18 * MB);  // 4 MB
    unsigned* msk  = (unsigned*)(ws + 22 * MB);  // 2 MB
    float*    yb   = (float*)(ws + 24 * MB);     // 8 MB

    // f32 -> f16 conversions
    int nx4 = NNODES * HID / 4;
    int nw4 = HID * HID / 4;
    cvt_f16_kernel<<<(nx4 + 255) / 256, 256, 0, stream>>>(x, xh, nx4);
    cvt_f16_kernel<<<(nw4 + 255) / 256, 256, 0, stream>>>(Wq, Whll + 0 * (size_t)HID * HID, nw4);
    cvt_f16_kernel<<<(nw4 + 255) / 256, 256, 0, stream>>>(Wk, Whll + 1 * (size_t)HID * HID, nw4);
    cvt_f16_kernel<<<(nw4 + 255) / 256, 256, 0, stream>>>(Wv, Whll + 2 * (size_t)HID * HID, nw4);
    cvt_f16_kernel<<<(nw4 + 255) / 256, 256, 0, stream>>>(Wo, Whll + 3 * (size_t)HID * HID, nw4);

    // adjacency bitmask
    int nwords = NNODES * MASKW;
    mask_zero_kernel<<<(nwords + 255) / 256, 256, 0, stream>>>(msk, nwords);
    mask_build_kernel<<<(E + 255) / 256, 256, 0, stream>>>(ei, E, msk);

    // Q,K,V projections
    qkv_kernel<<<dim3(NNODES / 16, 2, 3), 128, 0, stream>>>(xh, Whll, bq, bk, bv, Qh, Kh, Vt);

    // flash attention
    attn_kernel<<<dim3(NNODES / 64, NHEADS), 128, 0, stream>>>(Qh, Kh, Vt, msk, aO);

    // output projection + residual
    oproj_kernel<<<dim3(NNODES / 16, 2), 128, 0, stream>>>(
        aO, Whll + 3 * (size_t)HID * HID, bo, x, yb);

    // LayerNorm -> d_out
    ln_kernel<<<NNODES, 128, 0, stream>>>(yb, lng, lnb, (float*)d_out);
}
